// CostVolumeLayer_64931315581439
// MI455X (gfx1250) — matrix operand order repaired and checked by reading, not compiled, not run on hardware
//
#include <hip/hip_runtime.h>

typedef __attribute__((ext_vector_type(16))) _Float16 v16h;
typedef __attribute__((ext_vector_type(8)))  _Float16 h8;
typedef __attribute__((ext_vector_type(8)))  float    v8f;

#define B_      4
#define N1_     4096
#define N2_     4096
#define D_      64
#define K_      16
#define CIN     131
#define CIN_PAD 160     // pad 131 -> 5 chunks of 32 (f16 WMMA K=32)
#define CMID    128
#define COUT    128
#define H1STRIDE 136    // 128 + 8 halves pad

// ---------------------------------------------------------------------------
// Kernel 1: per query point, top-16 nearest neighbors in xyz2 + normalized
// inverse-distance weights. xyz2 (48KB/batch) cached in LDS; register top-16
// with cached current-max threshold (static indexing only -> stays in VGPRs).
// ---------------------------------------------------------------------------
__global__ __launch_bounds__(256) void topk_kernel(
    const float* __restrict__ xyz1, const float* __restrict__ xyz2,
    int* __restrict__ knn_idx, float* __restrict__ knn_w)
{
    __shared__ float x2s[N2_], y2s[N2_], z2s[N2_];
    const int b  = blockIdx.x >> 4;                       // 16 blocks per batch
    const int n1 = ((blockIdx.x & 15) << 8) + threadIdx.x;

    const float* xz2 = xyz2 + b * 3 * N2_;
    for (int i = threadIdx.x; i < N2_; i += 256) {
        x2s[i] = xz2[i];
        y2s[i] = xz2[N2_ + i];
        z2s[i] = xz2[2 * N2_ + i];
    }
    __syncthreads();

    const float px = xyz1[b * 3 * N1_ + n1];
    const float py = xyz1[b * 3 * N1_ + N1_ + n1];
    const float pz = xyz1[b * 3 * N1_ + 2 * N1_ + n1];

    float bd[K_]; int bi[K_];
#pragma unroll
    for (int j = 0; j < K_; ++j) { bd[j] = 3.402823466e+38f; bi[j] = 0; }
    float worst = 3.402823466e+38f; int wj = 0;

    for (int n2 = 0; n2 < N2_; ++n2) {
        float dx = px - x2s[n2];
        float dy = py - y2s[n2];
        float dz = pz - z2s[n2];
        float d  = fmaf(dx, dx, fmaf(dy, dy, dz * dz));
        if (d < worst) {
#pragma unroll
            for (int j = 0; j < K_; ++j)
                if (j == wj) { bd[j] = d; bi[j] = n2; }
            worst = bd[0]; wj = 0;
#pragma unroll
            for (int j = 1; j < K_; ++j)
                if (bd[j] > worst) { worst = bd[j]; wj = j; }
        }
    }

    float inv[K_]; float s = 0.f;
#pragma unroll
    for (int j = 0; j < K_; ++j) {
        inv[j] = 1.0f / fmaxf(sqrtf(bd[j]), 1e-10f);
        s += inv[j];
    }
    const float rs = 1.0f / s;
    const int base = (b * N1_ + n1) * K_;
#pragma unroll
    for (int j = 0; j < K_; ++j) {
        knn_idx[base + j] = bi[j];
        knn_w[base + j]   = inv[j] * rs;
    }
}

// assemble a v16h fragment from two 16B LDS reads (ds_load_b128 x2)
__device__ __forceinline__ v16h load_frag(const _Float16* p0, const _Float16* p1)
{
    h8 lo = *(const h8*)p0;
    h8 hi = *(const h8*)p1;
    v16h r;
#pragma unroll
    for (int i = 0; i < 8; ++i) { r[i] = lo[i]; r[8 + i] = hi[i]; }
    return r;
}

// ---------------------------------------------------------------------------
// Kernel 2: fused gather -> [131->128 leaky] -> [128->128 leaky] -> weighted
// k-reduction. One block = 16 query points = 16 WMMA M-tiles (16 rows each =
// the 16 neighbors of one point). 8 waves; each wave owns 2 M-tiles and
// processes them together so every B fragment (weights) loaded from LDS feeds
// TWO v_wmma ops -> halves LDS fragment bandwidth per WMMA.
// ---------------------------------------------------------------------------
__global__ __launch_bounds__(256) void cost_mlp_kernel(
    const float* __restrict__ xyz1, const float* __restrict__ xyz2,
    const float* __restrict__ points1, const float* __restrict__ points2,
    const float* __restrict__ W1, const float* __restrict__ b1,
    const float* __restrict__ W2, const float* __restrict__ b2,
    const int* __restrict__ knn_idx, const float* __restrict__ knn_w,
    float* __restrict__ out)
{
    extern __shared__ char smem_raw[];
    float*    b1s  = (float*)smem_raw;                     // 128 f32
    float*    b2s  = b1s + 128;                            // 128 f32
    float*    wl   = b2s + 128;                            // 256 f32 (per-row weight)
    _Float16* feat = (_Float16*)(wl + 256);                // 256 x 160
    _Float16* w1s  = feat + 256 * CIN_PAD;                 // 128 x 160  ([o][c])
    _Float16* w2s  = w1s + 128 * CIN_PAD;                  // 128 x 128  ([o2][o1])
    _Float16* h1s  = w2s + 128 * 128;                      // 8 waves x 32 x 136

    const int tid      = threadIdx.x;
    const int b        = blockIdx.x >> 8;                  // 256 blocks per batch
    const int n1_base  = (blockIdx.x & 255) << 4;          // 16 points per block

    // ---- stage weights (f16) + biases ----
    for (int i = tid; i < 128 * CIN_PAD; i += 256) {
        int o = i / CIN_PAD, c = i - o * CIN_PAD;
        w1s[i] = (_Float16)((c < CIN) ? W1[o * CIN + c] : 0.0f);
    }
    for (int i = tid; i < 128 * 128; i += 256)
        w2s[i] = (_Float16)W2[i];
    if (tid < 128) { b1s[tid] = b1[tid]; b2s[tid] = b2[tid]; }

    // ---- gather one feature row per thread: row = n1_local*16 + k ----
    {
        const int n1l = tid >> 4, k = tid & 15;
        const int n1  = n1_base + n1l;
        const int gbase = (b * N1_ + n1) * K_ + k;
        const int idx = knn_idx[gbase];
        wl[tid] = knn_w[gbase];

        _Float16* fr = feat + tid * CIN_PAD;
        const float* p1p = points1 + b * D_ * N1_ + n1;
        const float* p2p = points2 + b * D_ * N2_ + idx;
#pragma unroll
        for (int c = 0; c < D_; ++c) fr[c]       = (_Float16)p1p[c * N1_];
#pragma unroll
        for (int c = 0; c < D_; ++c) fr[D_ + c]  = (_Float16)p2p[c * N2_];
        const float* a1 = xyz1 + b * 3 * N1_ + n1;
        const float* a2 = xyz2 + b * 3 * N2_ + idx;
#pragma unroll
        for (int j = 0; j < 3; ++j)
            fr[2 * D_ + j] = (_Float16)(a2[j * N2_] - a1[j * N1_]);
#pragma unroll
        for (int c = CIN; c < CIN_PAD; ++c) fr[c] = (_Float16)0.0f;
    }
    __syncthreads();

    // ---- per-wave WMMA pipeline: 2 tiles per wave, shared B fragments ----
    const int wave = tid >> 5, lane = tid & 31;
    const int g = lane >> 4, n = lane & 15;                 // g: half-wave group
    const int t0 = wave * 2;                                // first of 2 tiles
    _Float16* h1w = h1s + wave * (32 * H1STRIDE);

    // hoist read-only scalars into registers (defeats smem alias reloads)
    float wv[2][8], b1v[8], b2v[8];
#pragma unroll
    for (int tt = 0; tt < 2; ++tt)
#pragma unroll
        for (int r = 0; r < 8; ++r)
            wv[tt][r] = wl[(t0 + tt) * 16 + r + 8 * g];
#pragma unroll
    for (int nt = 0; nt < 8; ++nt) {
        b1v[nt] = b1s[nt * 16 + n];
        b2v[nt] = b2s[nt * 16 + n];
    }

    const _Float16* frow0 = feat + (t0 * 16 + n) * CIN_PAD;  // A row m = lane&15
    const _Float16* frow1 = frow0 + 16 * CIN_PAD;

    // GEMM1: [16 x 160] x [160 x 128], 2 M-tiles
    v8f acc[2][8] = {};
#pragma unroll
    for (int kc = 0; kc < 5; ++kc) {
        const int ko = kc * 32 + g * 8;
        v16h a0 = load_frag(frow0 + ko, frow0 + ko + 16);
        v16h a1 = load_frag(frow1 + ko, frow1 + ko + 16);
#pragma unroll
        for (int nt = 0; nt < 8; ++nt) {
            const _Float16* q = w1s + (nt * 16 + n) * CIN_PAD + kc * 32 + g * 8;
            v16h bf = load_frag(q, q + 16);
            acc[0][nt] = __builtin_amdgcn_wmma_f32_16x16x32_f16(
                false, a0, false, bf, (short)0, acc[0][nt], false, false);
            acc[1][nt] = __builtin_amdgcn_wmma_f32_16x16x32_f16(
                false, a1, false, bf, (short)0, acc[1][nt], false, false);
        }
    }

    // epilogue 1: bias + leaky, restage as f16 A-layout rows in LDS
#pragma unroll
    for (int tt = 0; tt < 2; ++tt)
#pragma unroll
        for (int nt = 0; nt < 8; ++nt)
#pragma unroll
            for (int r = 0; r < 8; ++r) {
                const int m = r + 8 * g;                    // D-layout row
                float v = acc[tt][nt][r] + b1v[nt];
                v = v >= 0.f ? v : 0.1f * v;
                h1w[(tt * 16 + m) * H1STRIDE + nt * 16 + n] = (_Float16)v;
            }

    // GEMM2: [16 x 128] x [128 x 128], 2 M-tiles, shared B fragments
    v8f acc2[2][8] = {};
#pragma unroll
    for (int kc = 0; kc < 4; ++kc) {
        const int ko = kc * 32 + g * 8;
        const _Float16* h0 = h1w + n * H1STRIDE + ko;
        const _Float16* h1 = h0 + 16 * H1STRIDE;
        v16h a0 = load_frag(h0, h0 + 16);
        v16h a1 = load_frag(h1, h1 + 16);
#pragma unroll
        for (int nt = 0; nt < 8; ++nt) {
            const _Float16* q = w2s + (nt * 16 + n) * 128 + kc * 32 + g * 8;
            v16h bf = load_frag(q, q + 16);
            acc2[0][nt] = __builtin_amdgcn_wmma_f32_16x16x32_f16(
                false, a0, false, bf, (short)0, acc2[0][nt], false, false);
            acc2[1][nt] = __builtin_amdgcn_wmma_f32_16x16x32_f16(
                false, a1, false, bf, (short)0, acc2[1][nt], false, false);
        }
    }

    // epilogue 2: bias + leaky + inverse-distance weighted sum over the
    // 16 rows (= 16 neighbors of point t), then store out[b][o2][n1].
#pragma unroll
    for (int tt = 0; tt < 2; ++tt)
#pragma unroll
        for (int nt = 0; nt < 8; ++nt) {
            float part = 0.f;
#pragma unroll
            for (int r = 0; r < 8; ++r) {
                float v = acc2[tt][nt][r] + b2v[nt];
                v = v >= 0.f ? v : 0.1f * v;
                part += wv[tt][r] * v;
            }
            part += __shfl_xor(part, 16, 32);               // combine g=0/g=1 rows
            if (lane < 16) {
                const int n1 = n1_base + t0 + tt;
                out[(b * COUT + nt * 16 + n) * N1_ + n1] = part;
            }
        }
}

// ---------------------------------------------------------------------------
extern "C" void kernel_launch(void* const* d_in, const int* in_sizes, int n_in,
                              void* d_out, int out_size, void* d_ws, size_t ws_size,
                              hipStream_t stream)
{
    const float* xyz1    = (const float*)d_in[0];
    const float* xyz2    = (const float*)d_in[1];
    const float* points1 = (const float*)d_in[2];
    const float* points2 = (const float*)d_in[3];
    const float* W1      = (const float*)d_in[4];
    const float* b1      = (const float*)d_in[5];
    const float* W2      = (const float*)d_in[6];
    const float* b2      = (const float*)d_in[7];

    int*   knn_idx = (int*)d_ws;
    float* knn_w   = (float*)((char*)d_ws + (size_t)B_ * N1_ * K_ * sizeof(int));

    topk_kernel<<<dim3(B_ * (N1_ / 256)), dim3(256), 0, stream>>>(
        xyz1, xyz2, knn_idx, knn_w);

    const size_t shmem =
        (size_t)(128 + 128 + 256) * sizeof(float) +
        (size_t)(256 * CIN_PAD + 128 * CIN_PAD + 128 * 128 + 8 * 32 * H1STRIDE) *
            sizeof(_Float16);

    cost_mlp_kernel<<<dim3(B_ * (N1_ / 16)), dim3(256), shmem, stream>>>(
        xyz1, xyz2, points1, points2, W1, b1, W2, b2,
        knn_idx, knn_w, (float*)d_out);
}
